// DrugProteinGNN_27066883899924
// MI455X (gfx1250) — matrix operand order repaired and checked by reading, not compiled, not run on hardware
//
#include <hip/hip_runtime.h>

typedef __attribute__((ext_vector_type(16))) __bf16 v16bf;
typedef __attribute__((ext_vector_type(8)))  float  v8f;

#define Dm 128
#define BSZ 256
#define LSEQ 1000
#define NATOMS 10240
#define EDRUG 40960
#define NAFF 20000
#define EAFF 1000000
#define NUMDRUG 10000
#define EPSV 1e-5f
#define SLOPEV 0.01f
#define NEGV (-1e9f)
#define TEMP_INV 2.0f

__device__ __forceinline__ float leaky_f(float x){ return x >= 0.f ? x : SLOPEV * x; }
__device__ __forceinline__ unsigned short bf16bits(float f){
  union { __bf16 h; unsigned short s; } u; u.h = (__bf16)f; return u.s;
}

// CDNA5 async global->LDS copy (ASYNCcnt path), one b128 per enabled lane.
__device__ __forceinline__ void async_ld_b128(unsigned lds_byte_off, const void* gaddr){
  asm volatile("global_load_async_to_lds_b128 %0, %1, off"
               :: "v"(lds_byte_off), "v"(gaddr) : "memory");
}
__device__ __forceinline__ void wait_asynccnt0(){
#if __has_builtin(__builtin_amdgcn_s_wait_asynccnt)
  __builtin_amdgcn_s_wait_asynccnt(0);
#else
  asm volatile("s_wait_asynccnt 0x0" ::: "memory");
#endif
}

// ---------------- weight packing into WMMA B-fragment layout ----------------
// layout: idx = ((nt*(K/32)+kb)*32 + lane)*8 + v ; lane<16: N=lane, K pairs 2v,2v+1 (+kb*32)
//         lanes 16-31: K += 16
__global__ void pack_b_plain(const float* __restrict__ W, unsigned int* __restrict__ P,
                             int K, int N, int Ksrc){
  int tid = blockIdx.x * blockDim.x + threadIdx.x;
  int total = (N >> 4) * (K >> 5) * 256;
  if (tid >= total) return;
  int v = tid & 7, lane = (tid >> 3) & 31, blk = tid >> 8;
  int Kb = K >> 5;
  int kb = blk % Kb, nt = blk / Kb;
  int n  = nt * 16 + (lane & 15);
  int k0 = kb * 32 + ((lane >> 4) << 4) + (v << 1);
  float lo = (k0     < Ksrc) ? W[(size_t)k0 * N + n]       : 0.f;
  float hi = (k0 + 1 < Ksrc) ? W[(size_t)(k0 + 1) * N + n] : 0.f;
  P[tid] = ((unsigned)bf16bits(hi) << 16) | bf16bits(lo);
}

// conv weights w[co,ci,dk] ('OIH', [128,128,3]) -> B[k= dk*128+ci][n=co], K=384
__global__ void pack_b_conv(const float* __restrict__ w, unsigned int* __restrict__ P){
  int tid = blockIdx.x * blockDim.x + threadIdx.x;
  int total = 8 * 12 * 256;
  if (tid >= total) return;
  int v = tid & 7, lane = (tid >> 3) & 31, blk = tid >> 8;
  int kb = blk % 12, nt = blk / 12;
  int n  = nt * 16 + (lane & 15);
  int k0 = kb * 32 + ((lane >> 4) << 4) + (v << 1);
  float lo = w[((size_t)n * 128 + (k0 & 127)) * 3 + (k0 >> 7)];
  float hi = w[((size_t)n * 128 + ((k0 + 1) & 127)) * 3 + ((k0 + 1) >> 7)];
  P[tid] = ((unsigned)bf16bits(hi) << 16) | bf16bits(lo);
}

// B = S^T : B[k][n] = S[n*K + k]  (for sim = reps @ reps^T)
__global__ void pack_b_transpose(const float* __restrict__ S, unsigned int* __restrict__ P,
                                 int K, int N){
  int tid = blockIdx.x * blockDim.x + threadIdx.x;
  int total = (N >> 4) * (K >> 5) * 256;
  if (tid >= total) return;
  int v = tid & 7, lane = (tid >> 3) & 31, blk = tid >> 8;
  int Kb = K >> 5;
  int kb = blk % Kb, nt = blk / Kb;
  int n  = nt * 16 + (lane & 15);
  int k0 = kb * 32 + ((lane >> 4) << 4) + (v << 1);
  float lo = S[(size_t)n * K + k0];
  float hi = S[(size_t)n * K + k0 + 1];
  P[tid] = ((unsigned)bf16bits(hi) << 16) | bf16bits(lo);
}

// ---------------- generic WMMA GEMM: C[M,N] = act(A[M,K](bf16) @ Bpack + bias) --------
__global__ void gemm_wmma(const __bf16* __restrict__ A, const unsigned int* __restrict__ Bp,
                          const float* __restrict__ bias, float* __restrict__ C,
                          int M, int N, int K, int lda, int act){
  int wave = threadIdx.x >> 5;
  int lane = threadIdx.x & 31;
  int m0   = blockIdx.x << 4;
  int ntile = blockIdx.y * 8 + wave;
  int n0   = ntile << 4;
  int Kb   = K >> 5;
  int mrow = lane & 15;
  int khalf = (lane >> 4) << 4;
  const __bf16* Arow = A + (size_t)(m0 + mrow) * lda + khalf;
  const unsigned int* Bb = Bp + (size_t)ntile * Kb * 256 + lane * 8;
  union { v16bf v; unsigned int u[8]; } af, bfr;
  v8f acc = {};
  for (int kb = 0; kb < Kb; ++kb){
    const __bf16* ak = Arow + kb * 32;
#pragma unroll
    for (int v = 0; v < 8; ++v){
      int koff = v << 1;
      af.u[v] = *(const unsigned int*)(ak + koff);
    }
    const unsigned int* bk = Bb + (size_t)kb * 256;
#pragma unroll
    for (int v = 0; v < 8; ++v) bfr.u[v] = bk[v];
    acc = __builtin_amdgcn_wmma_f32_16x16x32_bf16(false, af.v, false, bfr.v,
                                                  (short)0, acc, false, false);
  }
  int n = n0 + (lane & 15);
  int mbase = m0 + ((lane >> 4) << 3);
  float bv = bias ? bias[n] : 0.f;
#pragma unroll
  for (int r = 0; r < 8; ++r){
    float v = acc[r] + bv;
    if (act) v = leaky_f(v);
    C[(size_t)(mbase + r) * N + n] = v;
  }
}

// ------------- implicit-im2col conv GEMM (k=3, pad=1) + BN stats accumulation ----------
// in/out layout: [b, l, c] (bf16).  K = 3*128 = 384, k = dk*128 + ci.
// Block: 64 output rows (4 M-subtiles) x 128 channels; A tile (66 rows halo) staged in
// LDS via CDNA5 async global->LDS loads; each wave reuses its B fragment 4x.
__global__ void conv_gemm(const __bf16* __restrict__ Pin, const unsigned int* __restrict__ Bp,
                          __bf16* __restrict__ Pout, float* __restrict__ chSum,
                          float* __restrict__ chSqs){
  __shared__ __align__(16) unsigned alds[66 * 64];  // 66 rows x 256B
  int b  = blockIdx.x >> 4;
  int l0 = (blockIdx.x & 15) << 6;  // 64-row tile
  const __bf16* base = Pin + (size_t)b * LSEQ * Dm;

  // stage rows l0-1 .. l0+64 into LDS (zero-fill halo / tail)
  for (int u = threadIdx.x; u < 66 * 16; u += 256){
    int row = u >> 4, seg = u & 15;            // 16B segment within 256B row
    int l = l0 - 1 + row;
    unsigned lds_off = (unsigned)(row * 256 + seg * 16);
    if (l >= 0 && l < LSEQ){
      async_ld_b128(lds_off + (unsigned)(size_t)alds,
                    (const char*)base + (size_t)l * 256 + seg * 16);
    } else {
      unsigned* p = &alds[row * 64 + seg * 4];
      p[0] = 0u; p[1] = 0u; p[2] = 0u; p[3] = 0u;
    }
  }
  wait_asynccnt0();
  __syncthreads();

  int wave = threadIdx.x >> 5;
  int lane = threadIdx.x & 31;
  int n0 = wave << 4;
  int mrow = lane & 15;
  int khalf = (lane >> 4) << 4;
  const unsigned int* Bb = Bp + (size_t)wave * 12 * 256 + lane * 8;
  union { v16bf v; unsigned int u[8]; } af, bfr;
  v8f acc0 = {}, acc1 = {}, acc2 = {}, acc3 = {};
  for (int kb = 0; kb < 12; ++kb){
    const unsigned int* bk = Bb + (size_t)kb * 256;
#pragma unroll
    for (int v = 0; v < 8; ++v) bfr.u[v] = bk[v];
#pragma unroll
    for (int ms = 0; ms < 4; ++ms){
#pragma unroll
      for (int v = 0; v < 8; ++v){
        int k = kb * 32 + khalf + (v << 1);
        int dk = k >> 7, ci = k & 127;
        int row = ms * 16 + mrow + dk;         // (l - (l0-1)) for tap dk
        af.u[v] = alds[row * 64 + (ci >> 1)];
      }
      v8f* accp = (ms == 0) ? &acc0 : (ms == 1) ? &acc1 : (ms == 2) ? &acc2 : &acc3;
      *accp = __builtin_amdgcn_wmma_f32_16x16x32_bf16(false, af.v, false, bfr.v,
                                                      (short)0, *accp, false, false);
    }
  }
  int n = n0 + (lane & 15);
  int lhalf = (lane >> 4) << 3;
  float s = 0.f, s2 = 0.f;
#pragma unroll
  for (int ms = 0; ms < 4; ++ms){
    v8f* accp = (ms == 0) ? &acc0 : (ms == 1) ? &acc1 : (ms == 2) ? &acc2 : &acc3;
#pragma unroll
    for (int r = 0; r < 8; ++r){
      int l = l0 + ms * 16 + lhalf + r;
      if (l < LSEQ){
        float v = (*accp)[r];
        Pout[((size_t)b * LSEQ + l) * Dm + n] = (__bf16)v;
        s += v; s2 += v * v;
      }
    }
  }
  atomicAdd(&chSum[n], s);
  atomicAdd(&chSqs[n], s2);
}

// ---------------- elementwise / GCN / pooling / loss kernels ----------------
__global__ void padbf_drugx(const float* __restrict__ x, __bf16* __restrict__ out){
  int i = blockIdx.x * blockDim.x + threadIdx.x;
  if (i >= NATOMS * 96) return;
  int k = i % 96, m = i / 96;
  out[i] = (__bf16)((k < 78) ? x[(size_t)m * 78 + k] : 0.f);
}

__global__ void f2bf_k(const float* __restrict__ in, __bf16* __restrict__ out, long long total){
  long long i = (long long)blockIdx.x * blockDim.x + threadIdx.x;
  if (i < total) out[i] = (__bf16)in[i];
}

__global__ void embed_k(const int* __restrict__ seq, const float* __restrict__ emb,
                        __bf16* __restrict__ out, long long total){
  long long i = (long long)blockIdx.x * blockDim.x + threadIdx.x;
  if (i >= total) return;
  int c = (int)(i & 127);
  long long bl = i >> 7;
  int tok = seq[bl];
  float v = (tok == 0) ? 0.f : emb[(size_t)tok * Dm + c];
  out[i] = (__bf16)v;
}

__global__ void deg_init(float* deg, int n){
  int i = blockIdx.x * blockDim.x + threadIdx.x;
  if (i < n) deg[i] = 1.f;                       // self-loop
}
__global__ void deg_accum(const int* __restrict__ ei, const float* __restrict__ ew,
                          float* deg, int E){
  int e = blockIdx.x * blockDim.x + threadIdx.x;
  if (e >= E) return;
  atomicAdd(&deg[ei[E + e]], ew ? ew[e] : 1.f);
}
__global__ void dinv_k(const float* __restrict__ deg, float* __restrict__ dinv, int n){
  int i = blockIdx.x * blockDim.x + threadIdx.x;
  if (i >= n) return;
  float d = deg[i];
  dinv[i] = d > 0.f ? rsqrtf(fmaxf(d, EPSV)) : 0.f;
}
__global__ void agg_init(const float* __restrict__ xw, const float* __restrict__ dinv,
                         float* __restrict__ out, int n){
  long long i = (long long)blockIdx.x * blockDim.x + threadIdx.x;
  if (i >= (long long)n * Dm) return;
  int node = (int)(i >> 7);
  float c = dinv[node];
  out[i] = c * c * xw[i];                        // self-loop term
}
__global__ void agg_edges(const int* __restrict__ ei, const float* __restrict__ ew,
                          const float* __restrict__ dinv, const float* __restrict__ xw,
                          float* __restrict__ out, int E){
  long long tid = (long long)blockIdx.x * blockDim.x + threadIdx.x;
  int e = (int)(tid >> 5);
  if (e >= E) return;
  int f = ((int)tid & 31) << 2;
  int r = ei[e], c = ei[E + e];
  float w = ew ? ew[e] : 1.f;
  float coef = dinv[r] * w * dinv[c];
  const float* src = xw + (size_t)r * Dm + f;
  float* dst = out + (size_t)c * Dm + f;
  atomicAdd(dst + 0, coef * src[0]);
  atomicAdd(dst + 1, coef * src[1]);
  atomicAdd(dst + 2, coef * src[2]);
  atomicAdd(dst + 3, coef * src[3]);
}
__global__ void bias_act_bf(const float* __restrict__ in, const float* __restrict__ bias,
                            float* __restrict__ f32out, __bf16* __restrict__ bfout,
                            long long total){
  long long i = (long long)blockIdx.x * blockDim.x + threadIdx.x;
  if (i >= total) return;
  int c = (int)(i & 127);
  float v = leaky_f(in[i] + bias[c]);
  if (f32out) f32out[i] = v;
  if (bfout)  bfout[i] = (__bf16)v;
}

__global__ void pool_accum(const float* __restrict__ h, const int* __restrict__ batch,
                           float* __restrict__ repsum, float* __restrict__ cnt, int n){
  long long tid = (long long)blockIdx.x * blockDim.x + threadIdx.x;
  int node = (int)(tid >> 5);
  if (node >= n) return;
  int f = ((int)tid & 31) << 2;
  int b = batch[node];
  const float* src = h + (size_t)node * Dm + f;
  float* dst = repsum + (size_t)b * Dm + f;
  atomicAdd(dst + 0, src[0]); atomicAdd(dst + 1, src[1]);
  atomicAdd(dst + 2, src[2]); atomicAdd(dst + 3, src[3]);
  if (f == 0) atomicAdd(&cnt[b], 1.f);
}
__global__ void pool_div(float* __restrict__ rep, const float* __restrict__ cnt){
  int i = blockIdx.x * blockDim.x + threadIdx.x;
  if (i >= BSZ * Dm) return;
  rep[i] /= fmaxf(cnt[i >> 7], 1.f);
}
__global__ void prot_pool(const __bf16* __restrict__ p, float* __restrict__ rep){
  int i = blockIdx.x * blockDim.x + threadIdx.x;
  if (i >= BSZ * Dm) return;
  int b = i >> 7, c = i & 127;
  float s = 0.f;
  for (int l = 0; l < LSEQ; ++l) s += (float)p[((size_t)b * LSEQ + l) * Dm + c];
  rep[i] = s * (1.f / LSEQ);
}

__global__ void bn_finalize(const float* __restrict__ sum, const float* __restrict__ sqs,
                            float* __restrict__ mean, float* __restrict__ rstd){
  int c = threadIdx.x;
  float inv = 1.f / (float)(BSZ * LSEQ);
  float m = sum[c] * inv;
  float v = sqs[c] * inv - m * m;
  mean[c] = m;
  rstd[c] = rsqrtf(v + EPSV);
}
__global__ void bn_apply(const __bf16* __restrict__ conv, const __bf16* __restrict__ resid,
                         const float* __restrict__ mean, const float* __restrict__ rstd,
                         const float* __restrict__ g, const float* __restrict__ beta,
                         __bf16* __restrict__ out, long long total){
  long long i = (long long)blockIdx.x * blockDim.x + threadIdx.x;
  if (i >= total) return;
  int c = (int)(i & 127);
  float x = (float)conv[i];
  float y = leaky_f((x - mean[c]) * rstd[c] * g[c] + beta[c]) + (float)resid[i];
  out[i] = (__bf16)y;
}

__global__ void gather_rows(const float* __restrict__ g, const int* __restrict__ ids,
                            int idoff, float* __restrict__ out){
  int i = blockIdx.x * blockDim.x + threadIdx.x;
  if (i >= BSZ * Dm) return;
  int b = i >> 7, c = i & 127;
  out[i] = g[((size_t)(ids[b] + idoff)) * Dm + c];
}

__global__ void norm_stack(const float* __restrict__ z1, const float* __restrict__ z2,
                           float* __restrict__ out){
  __shared__ float red[Dm];
  int row = blockIdx.x;
  const float* src = (row < BSZ) ? z1 + (size_t)row * Dm : z2 + (size_t)(row - BSZ) * Dm;
  float v = src[threadIdx.x];
  red[threadIdx.x] = v * v;
  __syncthreads();
  for (int s = 64; s > 0; s >>= 1){
    if (threadIdx.x < s) red[threadIdx.x] += red[threadIdx.x + s];
    __syncthreads();
  }
  float nrm = fmaxf(sqrtf(red[0]), 1e-12f);
  out[(size_t)row * Dm + threadIdx.x] = v / nrm;
}

__global__ void simclr_loss(const float* __restrict__ sim, float* __restrict__ lossOut){
  __shared__ float sm[256];
  int i = blockIdx.x;
  float mx = -3.4e38f;
  for (int j = threadIdx.x; j < 512; j += 256){
    float v = (j == i) ? NEGV : sim[(size_t)i * 512 + j] * TEMP_INV;
    mx = fmaxf(mx, v);
  }
  sm[threadIdx.x] = mx; __syncthreads();
  for (int s = 128; s > 0; s >>= 1){
    if (threadIdx.x < s) sm[threadIdx.x] = fmaxf(sm[threadIdx.x], sm[threadIdx.x + s]);
    __syncthreads();
  }
  float m = sm[0]; __syncthreads();
  float se = 0.f;
  for (int j = threadIdx.x; j < 512; j += 256){
    float v = (j == i) ? NEGV : sim[(size_t)i * 512 + j] * TEMP_INV;
    se += __expf(v - m);
  }
  sm[threadIdx.x] = se; __syncthreads();
  for (int s = 128; s > 0; s >>= 1){
    if (threadIdx.x < s) sm[threadIdx.x] += sm[threadIdx.x + s];
    __syncthreads();
  }
  if (threadIdx.x == 0){
    int lbl = (i < BSZ) ? i + BSZ : i - BSZ;
    float vl = sim[(size_t)i * 512 + lbl] * TEMP_INV;
    float logp = vl - (m + __logf(sm[0]));
    atomicAdd(lossOut, -logp * (1.f / 512.f));
  }
}

__global__ void dp_concat(const float* __restrict__ dr, const float* __restrict__ dg,
                          const float* __restrict__ pr, const float* __restrict__ pg,
                          __bf16* __restrict__ dp){
  int i = blockIdx.x * blockDim.x + threadIdx.x;
  if (i >= BSZ * 512) return;
  int b = i >> 9, c = i & 511;
  float v;
  if      (c < 128) v = dr[(size_t)b * Dm + c];
  else if (c < 256) v = dg[(size_t)b * Dm + c - 128];
  else if (c < 384) v = pr[(size_t)b * Dm + c - 256];
  else              v = pg[(size_t)b * Dm + c - 384];
  dp[i] = (__bf16)v;
}

__global__ void head_y(const float* __restrict__ hcls, const float* __restrict__ Wp2,
                       const float* __restrict__ bp2, float* __restrict__ y){
  int b = blockIdx.x * blockDim.x + threadIdx.x;
  if (b >= BSZ) return;
  float s = bp2[0];
  for (int c = 0; c < Dm; ++c) s += hcls[(size_t)b * Dm + c] * Wp2[c];
  y[b] = s;
}

// --------------------------------- host ---------------------------------
static inline int cdiv(long long a, long long b){ return (int)((a + b - 1) / b); }

extern "C" void kernel_launch(void* const* d_in, const int* in_sizes, int n_in,
                              void* d_out, int out_size, void* d_ws, size_t ws_size,
                              hipStream_t stream) {
  const float* drug_x   = (const float*)d_in[0];
  const int*   edge_idx = (const int*)d_in[1];
  const int*   batch    = (const int*)d_in[2];
  const int*   prot_seq = (const int*)d_in[3];
  const int*   drug_ids = (const int*)d_in[4];
  const int*   prot_ids = (const int*)d_in[5];
  const float* aff_x    = (const float*)d_in[6];
  const int*   aff_ei   = (const int*)d_in[7];
  const float* aff_ew   = (const float*)d_in[8];
  const float* Wd1 = (const float*)d_in[9];  const float* bd1 = (const float*)d_in[10];
  const float* Wd2 = (const float*)d_in[11]; const float* bd2 = (const float*)d_in[12];
  const float* emb = (const float*)d_in[13];
  const float* c1w = (const float*)d_in[14];
  const float* bn1g = (const float*)d_in[16]; const float* bn1b = (const float*)d_in[17];
  const float* c2w = (const float*)d_in[18];
  const float* bn2g = (const float*)d_in[20]; const float* bn2b = (const float*)d_in[21];
  const float* Wg1 = (const float*)d_in[22]; const float* bg1 = (const float*)d_in[23];
  const float* Wg2 = (const float*)d_in[24]; const float* bg2 = (const float*)d_in[25];
  const float* Wp1 = (const float*)d_in[26]; const float* bp1 = (const float*)d_in[27];
  const float* Wp2 = (const float*)d_in[28]; const float* bp2 = (const float*)d_in[29];
  float* out = (float*)d_out;

  char* ws = (char*)d_ws;
  size_t off = 0;
  auto alloc = [&](size_t b){ size_t o = off; off += (b + 255) & ~(size_t)255; return o; };
  size_t o_Adrug  = alloc((size_t)NATOMS * 96 * 2);
  size_t o_Wd1p   = alloc(8 * 3 * 256 * 4);
  size_t o_Wd2p   = alloc(8 * 4 * 256 * 4);
  size_t o_Wg1p   = alloc(8 * 4 * 256 * 4);
  size_t o_Wg2p   = alloc(8 * 4 * 256 * 4);
  size_t o_Wp1p   = alloc(8 * 16 * 256 * 4);
  size_t o_C1p    = alloc(8 * 12 * 256 * 4);
  size_t o_C2p    = alloc(8 * 12 * 256 * 4);
  size_t o_affbf  = alloc((size_t)NAFF * Dm * 2);
  size_t o_hbf    = alloc((size_t)NAFF * Dm * 2);
  size_t o_xw     = alloc((size_t)NAFF * Dm * 4);
  size_t o_agg    = alloc((size_t)NAFF * Dm * 4);
  size_t o_g      = alloc((size_t)NAFF * Dm * 4);
  size_t o_deg    = alloc((size_t)NAFF * 4);
  size_t o_dinv   = alloc((size_t)NAFF * 4);
  size_t o_drep   = alloc((size_t)BSZ * Dm * 4);
  size_t o_cnt    = alloc((size_t)BSZ * 4);
  size_t o_prep   = alloc((size_t)BSZ * Dm * 4);
  size_t o_dg     = alloc((size_t)BSZ * Dm * 4);
  size_t o_pg     = alloc((size_t)BSZ * Dm * 4);
  size_t o_p0     = alloc((size_t)BSZ * LSEQ * Dm * 2);
  size_t o_p1     = alloc((size_t)BSZ * LSEQ * Dm * 2);
  size_t o_conv   = alloc((size_t)BSZ * LSEQ * Dm * 2);
  size_t o_chst   = alloc(4 * Dm * 4);          // sum | sqs | mean | rstd
  size_t o_repsN  = alloc((size_t)512 * Dm * 4);
  size_t o_repsA  = alloc((size_t)512 * Dm * 2);
  size_t o_repsBp = alloc(32 * 4 * 256 * 4);
  size_t o_sim    = alloc((size_t)512 * 512 * 4);
  size_t o_dp     = alloc((size_t)BSZ * 512 * 2);
  size_t o_hcls   = alloc((size_t)BSZ * Dm * 4);

  __bf16* Adrug = (__bf16*)(ws + o_Adrug);
  unsigned int* Wd1p = (unsigned int*)(ws + o_Wd1p);
  unsigned int* Wd2p = (unsigned int*)(ws + o_Wd2p);
  unsigned int* Wg1p = (unsigned int*)(ws + o_Wg1p);
  unsigned int* Wg2p = (unsigned int*)(ws + o_Wg2p);
  unsigned int* Wp1p = (unsigned int*)(ws + o_Wp1p);
  unsigned int* C1p  = (unsigned int*)(ws + o_C1p);
  unsigned int* C2p  = (unsigned int*)(ws + o_C2p);
  __bf16* affbf = (__bf16*)(ws + o_affbf);
  __bf16* hbf   = (__bf16*)(ws + o_hbf);
  float* xw   = (float*)(ws + o_xw);
  float* agg  = (float*)(ws + o_agg);
  float* gbuf = (float*)(ws + o_g);
  float* deg  = (float*)(ws + o_deg);
  float* dinv = (float*)(ws + o_dinv);
  float* drep = (float*)(ws + o_drep);
  float* cnt  = (float*)(ws + o_cnt);
  float* prep = (float*)(ws + o_prep);
  float* dgE  = (float*)(ws + o_dg);
  float* pgE  = (float*)(ws + o_pg);
  __bf16* p0   = (__bf16*)(ws + o_p0);
  __bf16* p1   = (__bf16*)(ws + o_p1);
  __bf16* conv = (__bf16*)(ws + o_conv);
  float* chSum  = (float*)(ws + o_chst);
  float* chSqs  = chSum + Dm;
  float* chMean = chSum + 2 * Dm;
  float* chRstd = chSum + 3 * Dm;
  float* repsN = (float*)(ws + o_repsN);
  __bf16* repsA = (__bf16*)(ws + o_repsA);
  unsigned int* repsBp = (unsigned int*)(ws + o_repsBp);
  float* sim  = (float*)(ws + o_sim);
  __bf16* dp  = (__bf16*)(ws + o_dp);
  float* hcls = (float*)(ws + o_hcls);

  const int T = 256;
  // ---- phase 0: conversions / weight packing ----
  padbf_drugx<<<cdiv((long long)NATOMS * 96, T), T, 0, stream>>>(drug_x, Adrug);
  pack_b_plain<<<cdiv(8 * 3 * 256, T), T, 0, stream>>>(Wd1, Wd1p, 96, 128, 78);
  pack_b_plain<<<cdiv(8 * 4 * 256, T), T, 0, stream>>>(Wd2, Wd2p, 128, 128, 128);
  pack_b_plain<<<cdiv(8 * 4 * 256, T), T, 0, stream>>>(Wg1, Wg1p, 128, 128, 128);
  pack_b_plain<<<cdiv(8 * 4 * 256, T), T, 0, stream>>>(Wg2, Wg2p, 128, 128, 128);
  pack_b_plain<<<cdiv(8 * 16 * 256, T), T, 0, stream>>>(Wp1, Wp1p, 512, 128, 512);
  pack_b_conv<<<cdiv(8 * 12 * 256, T), T, 0, stream>>>(c1w, C1p);
  pack_b_conv<<<cdiv(8 * 12 * 256, T), T, 0, stream>>>(c2w, C2p);
  f2bf_k<<<cdiv((long long)NAFF * Dm, T), T, 0, stream>>>(aff_x, affbf, (long long)NAFF * Dm);
  embed_k<<<cdiv((long long)BSZ * LSEQ * Dm, T), T, 0, stream>>>(prot_seq, emb, p0,
                                                                 (long long)BSZ * LSEQ * Dm);
  // ---- phase 1: drug GCN (2 layers) + mean pool ----
  deg_init<<<cdiv(NATOMS, T), T, 0, stream>>>(deg, NATOMS);
  deg_accum<<<cdiv(EDRUG, T), T, 0, stream>>>(edge_idx, nullptr, deg, EDRUG);
  dinv_k<<<cdiv(NATOMS, T), T, 0, stream>>>(deg, dinv, NATOMS);
  gemm_wmma<<<dim3(NATOMS / 16, 1), T, 0, stream>>>(Adrug, Wd1p, nullptr, xw,
                                                    NATOMS, Dm, 96, 96, 0);
  agg_init<<<cdiv((long long)NATOMS * Dm, T), T, 0, stream>>>(xw, dinv, agg, NATOMS);
  agg_edges<<<cdiv((long long)EDRUG * 32, T), T, 0, stream>>>(edge_idx, nullptr, dinv, xw, agg, EDRUG);
  bias_act_bf<<<cdiv((long long)NATOMS * Dm, T), T, 0, stream>>>(agg, bd1, nullptr, hbf,
                                                                 (long long)NATOMS * Dm);
  gemm_wmma<<<dim3(NATOMS / 16, 1), T, 0, stream>>>(hbf, Wd2p, nullptr, xw,
                                                    NATOMS, Dm, 128, 128, 0);
  agg_init<<<cdiv((long long)NATOMS * Dm, T), T, 0, stream>>>(xw, dinv, agg, NATOMS);
  agg_edges<<<cdiv((long long)EDRUG * 32, T), T, 0, stream>>>(edge_idx, nullptr, dinv, xw, agg, EDRUG);
  bias_act_bf<<<cdiv((long long)NATOMS * Dm, T), T, 0, stream>>>(agg, bd2, gbuf, nullptr,
                                                                 (long long)NATOMS * Dm);
  hipMemsetAsync(drep, 0, (size_t)BSZ * Dm * 4, stream);
  hipMemsetAsync(cnt, 0, (size_t)BSZ * 4, stream);
  pool_accum<<<cdiv((long long)NATOMS * 32, T), T, 0, stream>>>(gbuf, batch, drep, cnt, NATOMS);
  pool_div<<<cdiv(BSZ * Dm, T), T, 0, stream>>>(drep, cnt);
  // ---- phase 2: protein conv encoder (LDS/async staged WMMA conv) ----
  long long ptot = (long long)BSZ * LSEQ * Dm;
  hipMemsetAsync(chSum, 0, 2 * Dm * 4, stream);
  conv_gemm<<<BSZ * 16, T, 0, stream>>>(p0, C1p, conv, chSum, chSqs);
  bn_finalize<<<1, Dm, 0, stream>>>(chSum, chSqs, chMean, chRstd);
  bn_apply<<<cdiv(ptot, T), T, 0, stream>>>(conv, p0, chMean, chRstd, bn1g, bn1b, p1, ptot);
  hipMemsetAsync(chSum, 0, 2 * Dm * 4, stream);
  conv_gemm<<<BSZ * 16, T, 0, stream>>>(p1, C2p, conv, chSum, chSqs);
  bn_finalize<<<1, Dm, 0, stream>>>(chSum, chSqs, chMean, chRstd);
  bn_apply<<<cdiv(ptot, T), T, 0, stream>>>(conv, p1, chMean, chRstd, bn2g, bn2b, p0, ptot);
  prot_pool<<<cdiv(BSZ * Dm, T), T, 0, stream>>>(p0, prep);
  // ---- phase 3: affinity GCN (2 layers) + gathers ----
  deg_init<<<cdiv(NAFF, T), T, 0, stream>>>(deg, NAFF);
  deg_accum<<<cdiv(EAFF, T), T, 0, stream>>>(aff_ei, aff_ew, deg, EAFF);
  dinv_k<<<cdiv(NAFF, T), T, 0, stream>>>(deg, dinv, NAFF);
  gemm_wmma<<<dim3(NAFF / 16, 1), T, 0, stream>>>(affbf, Wg1p, nullptr, xw, NAFF, Dm, 128, 128, 0);
  agg_init<<<cdiv((long long)NAFF * Dm, T), T, 0, stream>>>(xw, dinv, agg, NAFF);
  agg_edges<<<cdiv((long long)EAFF * 32, T), T, 0, stream>>>(aff_ei, aff_ew, dinv, xw, agg, EAFF);
  bias_act_bf<<<cdiv((long long)NAFF * Dm, T), T, 0, stream>>>(agg, bg1, nullptr, hbf,
                                                               (long long)NAFF * Dm);
  gemm_wmma<<<dim3(NAFF / 16, 1), T, 0, stream>>>(hbf, Wg2p, nullptr, xw, NAFF, Dm, 128, 128, 0);
  agg_init<<<cdiv((long long)NAFF * Dm, T), T, 0, stream>>>(xw, dinv, agg, NAFF);
  agg_edges<<<cdiv((long long)EAFF * 32, T), T, 0, stream>>>(aff_ei, aff_ew, dinv, xw, agg, EAFF);
  bias_act_bf<<<cdiv((long long)NAFF * Dm, T), T, 0, stream>>>(agg, bg2, gbuf, nullptr,
                                                               (long long)NAFF * Dm);
  gather_rows<<<cdiv(BSZ * Dm, T), T, 0, stream>>>(gbuf, drug_ids, 0, dgE);
  gather_rows<<<cdiv(BSZ * Dm, T), T, 0, stream>>>(gbuf, prot_ids, NUMDRUG, pgE);
  // ---- phase 4: SimCLR losses (WMMA similarity GEMMs) ----
  hipMemsetAsync(out + 256, 0, 2 * sizeof(float), stream);
  // drug loss
  norm_stack<<<512, Dm, 0, stream>>>(drep, dgE, repsN);
  f2bf_k<<<cdiv((long long)512 * Dm, T), T, 0, stream>>>(repsN, repsA, (long long)512 * Dm);
  pack_b_transpose<<<cdiv(32 * 4 * 256, T), T, 0, stream>>>(repsN, repsBp, 128, 512);
  gemm_wmma<<<dim3(32, 4), T, 0, stream>>>(repsA, repsBp, nullptr, sim, 512, 512, 128, 128, 0);
  simclr_loss<<<512, 256, 0, stream>>>(sim, out + 256);
  // protein loss
  norm_stack<<<512, Dm, 0, stream>>>(prep, pgE, repsN);
  f2bf_k<<<cdiv((long long)512 * Dm, T), T, 0, stream>>>(repsN, repsA, (long long)512 * Dm);
  pack_b_transpose<<<cdiv(32 * 4 * 256, T), T, 0, stream>>>(repsN, repsBp, 128, 512);
  gemm_wmma<<<dim3(32, 4), T, 0, stream>>>(repsA, repsBp, nullptr, sim, 512, 512, 128, 128, 0);
  simclr_loss<<<512, 256, 0, stream>>>(sim, out + 257);
  // ---- phase 5: prediction head ----
  dp_concat<<<cdiv(BSZ * 512, T), T, 0, stream>>>(drep, dgE, prep, pgE, dp);
  gemm_wmma<<<dim3(BSZ / 16, 1), T, 0, stream>>>(dp, Wp1p, bp1, hcls, BSZ, Dm, 512, 512, 1);
  head_y<<<1, 256, 0, stream>>>(hcls, Wp2, bp2, out);
  (void)in_sizes; (void)n_in; (void)out_size; (void)ws_size;
}
